// AttentionConv3D_31464930410846
// MI455X (gfx1250) — matrix operand order repaired and checked by reading, not compiled, not run on hardware
//
#include <hip/hip_runtime.h>

// ---------------------------------------------------------------------------
// AttentionConv3D (XCA / transposed channel attention) for gfx1250 (MI455X)
// Matrix math on v_wmma_f32_16x16x32_bf16 (wave32 WMMA); async LDS loads where
// the toolchain exposes them.
// ---------------------------------------------------------------------------

typedef __attribute__((ext_vector_type(16))) __bf16 v16bf;
typedef __attribute__((ext_vector_type(8)))  float  v8f;

#define HW      16384      // 128*128
#define BATCH   8
#define CDIM    256
#define QKV_CH  768
#define NHEADS  8

#if __has_builtin(__builtin_amdgcn_global_load_async_to_lds_b128)
#define HAVE_ASYNC_LDS 1
// exact parameter types leaked by clang's diagnostic:
typedef int v4i_vs __attribute__((vector_size(16)));
typedef v4i_vs __attribute__((address_space(1)))* g_v4i_p;  // prints as "__device__"
typedef v4i_vs __attribute__((address_space(3)))* l_v4i_p;  // prints as "__shared__"
#else
#define HAVE_ASYNC_LDS 0
#endif

__device__ __forceinline__ void wait_asynccnt0() {
#if __has_builtin(__builtin_amdgcn_s_wait_asynccnt)
  __builtin_amdgcn_s_wait_asynccnt(0);
#else
  asm volatile("s_wait_asynccnt 0x0" ::: "memory");
#endif
}

// ---- bf16 helpers (RNE) ----------------------------------------------------
__device__ __forceinline__ __bf16 f2b(float f) {
  unsigned u = __builtin_bit_cast(unsigned, f);
  u += 0x7FFFu + ((u >> 16) & 1u);
  unsigned short h = (unsigned short)(u >> 16);
  return __builtin_bit_cast(__bf16, h);
}
__device__ __forceinline__ float b2f(__bf16 b) {
  unsigned short h = __builtin_bit_cast(unsigned short, b);
  unsigned u = ((unsigned)h) << 16;
  return __builtin_bit_cast(float, u);
}
__device__ __forceinline__ void put2(v16bf& d, int v, unsigned u) {
  d[2 * v]     = __builtin_bit_cast(__bf16, (unsigned short)(u & 0xFFFFu));
  d[2 * v + 1] = __builtin_bit_cast(__bf16, (unsigned short)(u >> 16));
}

// A fragment (16x32 bf16, M rows across lanes, contiguous K matching B):
// lane half 0 -> K 0..15, half 1 -> K 16..31; rowp = K0 of row.
__device__ __forceinline__ v16bf load_afrag(const __bf16* rowp, int half) {
  const unsigned* p = (const unsigned*)rowp;   // rows kept 4B aligned
  v16bf a;
#pragma unroll
  for (int v = 0; v < 8; ++v) put2(a, v, p[half * 8 + v]);
  return a;
}
// B fragment from row=N, contiguous-K storage (lanes 0-15 K=0..15, 16-31 K=16..31).
__device__ __forceinline__ v16bf load_bfrag(const __bf16* rowp, int half) {
  const unsigned* p = (const unsigned*)rowp;
  v16bf b;
#pragma unroll
  for (int v = 0; v < 8; ++v) put2(b, v, p[half * 8 + v]);
  return b;
}
// B fragment from [k][n] LDS tile (stride 136 = 16B-aligned rows), strided gather.
__device__ __forceinline__ v16bf load_bfrag_kn(const __bf16* base, int N, int half) {
  v16bf b;
#pragma unroll
  for (int v = 0; v < 8; ++v) {
    int k = half * 16 + 2 * v;
    b[2 * v]     = base[k * 136 + N];
    b[2 * v + 1] = base[(k + 1) * 136 + N];
  }
  return b;
}

// ---------------------------------------------------------------------------
// Kernels 1/4: channel-mix GEMM  O[b,m,n] = sum_k A[m,k] * B[b,k,n]
// 128x128 block tile, BK=32, 8 waves (4x2), each wave 32x64 -> 2x4 WMMA frags.
// B_BF16 path keeps B LDS tile in [k][n] so it can be filled with
// GLOBAL_LOAD_ASYNC_TO_LDS_B128 (no VGPR round-trip, ASYNCcnt tracked).
// ---------------------------------------------------------------------------
template <bool B_BF16, bool OUT_BF16>
__global__ __launch_bounds__(256) void gemm_channelmix(
    const float* __restrict__ A,   // [M][K] f32 weights
    const void*  __restrict__ Bv,  // [BATCH][K][HW] f32 or bf16
    void*        __restrict__ Ov,  // [BATCH][M][HW] f32 or bf16
    int M, int K) {
  __shared__ __attribute__((aligned(16))) __bf16 Al[128 * 34];  // [m][k]
  __shared__ __attribute__((aligned(16))) __bf16 Bl[128 * 34];  // [n][k]@34 or [k][n]@136

  const int t    = threadIdx.x;
  const int lane = t & 31;
  const int half = lane >> 4;
  const int l15  = lane & 15;
  const int w    = t >> 5;
  const int wm   = w >> 1;   // 0..3
  const int wn   = w & 1;    // 0..1

  const int mBase = blockIdx.y * 128;
  const int nBase = blockIdx.x * 128;
  const int b     = blockIdx.z;

  const __bf16* Bb = (const __bf16*)Bv + (size_t)b * (size_t)K * HW;
  const float*  Bf = (const float*)Bv  + (size_t)b * (size_t)K * HW;

  v8f acc[2][4];
#pragma unroll
  for (int mf = 0; mf < 2; ++mf)
#pragma unroll
    for (int nf = 0; nf < 4; ++nf)
#pragma unroll
      for (int i = 0; i < 8; ++i) acc[mf][nf][i] = 0.0f;

  const int arow = t >> 1, acb = (t & 1) * 16;   // A-tile cooperative load map
  const int bk   = t >> 3, bnb = (t & 7) * 16;   // B-tile cooperative load map

  for (int k0 = 0; k0 < K; k0 += 32) {
    // A tile: 128 x 32, f32 -> bf16, row-major [m][k]
    {
      const float* src = A + (size_t)(mBase + arow) * K + k0 + acb;
      __bf16* dst = Al + arow * 34 + acb;
#pragma unroll
      for (int j = 0; j < 16; ++j) dst[j] = f2b(src[j]);
    }
    // B tile
    if (B_BF16) {
      // [k][n] layout, stride 136 (rows 16B aligned for async b128 stores)
      const __bf16* src = Bb + (size_t)(k0 + bk) * HW + nBase + bnb;
      __bf16* dst = &Bl[bk * 136 + bnb];
#if HAVE_ASYNC_LDS
      __builtin_amdgcn_global_load_async_to_lds_b128(
          (g_v4i_p)const_cast<__bf16*>(src), (l_v4i_p)dst, 0, 0);
      __builtin_amdgcn_global_load_async_to_lds_b128(
          (g_v4i_p)const_cast<__bf16*>(src + 8), (l_v4i_p)(dst + 8), 0, 0);
      wait_asynccnt0();
#else
#pragma unroll
      for (int j = 0; j < 16; ++j) dst[j] = src[j];
#endif
    } else {
      // f32 source: convert + transpose into [n][k], stride 34
      const float* src = Bf + (size_t)(k0 + bk) * HW + nBase + bnb;
#pragma unroll
      for (int j = 0; j < 16; ++j) Bl[(bnb + j) * 34 + bk] = f2b(src[j]);
    }
    // hint the next K-tile into cache (global_prefetch_b8)
    if (k0 + 32 < K) {
      if (B_BF16) __builtin_prefetch(Bb + (size_t)(k0 + 32 + bk) * HW + nBase + bnb, 0, 1);
      else        __builtin_prefetch(Bf + (size_t)(k0 + 32 + bk) * HW + nBase + bnb, 0, 1);
      __builtin_prefetch(A + (size_t)(mBase + arow) * K + k0 + 32 + acb, 0, 1);
    }
    __syncthreads();

    v16bf af[2], bfr[4];
#pragma unroll
    for (int mf = 0; mf < 2; ++mf)
      af[mf] = load_afrag(Al + (wm * 32 + mf * 16 + l15) * 34, half);
#pragma unroll
    for (int nf = 0; nf < 4; ++nf) {
      if (B_BF16) bfr[nf] = load_bfrag_kn(Bl, wn * 64 + nf * 16 + l15, half);
      else        bfr[nf] = load_bfrag(Bl + (wn * 64 + nf * 16 + l15) * 34, half);
    }
#pragma unroll
    for (int mf = 0; mf < 2; ++mf)
#pragma unroll
      for (int nf = 0; nf < 4; ++nf)
        acc[mf][nf] = __builtin_amdgcn_wmma_f32_16x16x32_bf16(
            false, af[mf], false, bfr[nf], (short)0, acc[mf][nf], false, false);
    __syncthreads();
  }

  // Epilogue: C/D layout -> VGPR r: M = r + 8*half, N = l15
#pragma unroll
  for (int mf = 0; mf < 2; ++mf)
#pragma unroll
    for (int nf = 0; nf < 4; ++nf)
#pragma unroll
      for (int r = 0; r < 8; ++r) {
        int gm = mBase + wm * 32 + mf * 16 + r + half * 8;
        int gn = nBase + wn * 64 + nf * 16 + l15;
        size_t o = ((size_t)b * M + gm) * HW + gn;
        float val = acc[mf][nf][r];
        if (OUT_BF16) ((__bf16*)Ov)[o] = f2b(val);
        else          ((float*)Ov)[o]  = val;
      }
}

// ---------------------------------------------------------------------------
// Kernel 2: depthwise 3x3 (middle slice of the 3x3x3 filter), SAME zero pad.
// ---------------------------------------------------------------------------
__global__ __launch_bounds__(256) void dwconv3x3(
    const __bf16* __restrict__ in,   // [BATCH][768][128][128]
    const float*  __restrict__ wdw,  // [768][1][3][3][3]
    __bf16* __restrict__ out) {
  int idx = blockIdx.x * 256 + threadIdx.x;
  int xx  = idx & 127;
  int yy  = (idx >> 7) & 127;
  int bc  = idx >> 14;                 // 0 .. BATCH*768-1
  int c   = bc % QKV_CH;
  const __bf16* p  = in + (size_t)bc * HW;
  const float*  wt = wdw + c * 27 + 9;  // [c][0][1][dy][dx]
  float s = 0.f;
#pragma unroll
  for (int dy = -1; dy <= 1; ++dy) {
    int y = yy + dy;
    if ((unsigned)y >= 128u) continue;
#pragma unroll
    for (int dx = -1; dx <= 1; ++dx) {
      int x = xx + dx;
      if ((unsigned)x >= 128u) continue;
      s += wt[(dy + 1) * 3 + (dx + 1)] * b2f(p[y * 128 + x]);
    }
  }
  out[(size_t)bc * HW + yy * 128 + xx] = f2b(s);
}

// ---------------------------------------------------------------------------
// Kernel Z: zero f32 reduction workspace (required every launch: graph replay).
// ---------------------------------------------------------------------------
__global__ __launch_bounds__(256) void zero_f32(float* __restrict__ p, int n) {
  int i = blockIdx.x * 256 + threadIdx.x;
  if (i < n) p[i] = 0.f;
}

// ---------------------------------------------------------------------------
// Kernel 3a: gram matrix G = q k^T + row sum-of-squares, partial over an
// n-slice; 512 blocks (8 slices x 64 (b,head)); LDS reduce -> global atomics.
// l2norm is a per-row scale, so norms fold into softmax later.
// ---------------------------------------------------------------------------
__global__ __launch_bounds__(256) void xca_gram(
    const __bf16* __restrict__ qkvc,  // [BATCH][768][HW]
    float* __restrict__ Gws,          // [64][32][32]
    float* __restrict__ sqws) {       // [64][64]  (q norms then k norms)
  __shared__ float Gs[32][32];
  __shared__ float sqq[32], sqk[32];

  const int t    = threadIdx.x;
  const int lane = t & 31;
  const int half = lane >> 4;
  const int l15  = lane & 15;
  const int w    = t >> 5;
  const int bh    = blockIdx.y;    // 0..63
  const int b     = bh >> 3;
  const int head  = bh & 7;
  const int slice = blockIdx.x;    // 0..7

  const __bf16* qb = qkvc + ((size_t)b * QKV_CH + head * 32) * HW;
  const __bf16* kb = qb + (size_t)256 * HW;

  for (int i = t; i < 1024; i += 256) ((float*)Gs)[i] = 0.f;
  if (t < 32) { sqq[t] = 0.f; sqk[t] = 0.f; }
  __syncthreads();

  v8f g[2][2];
#pragma unroll
  for (int mf = 0; mf < 2; ++mf)
#pragma unroll
    for (int nf = 0; nf < 2; ++nf)
#pragma unroll
      for (int i = 0; i < 8; ++i) g[mf][nf][i] = 0.f;
  float msq[2] = {0.f, 0.f}, msk[2] = {0.f, 0.f};

  const int it0 = slice * 64;
  for (int it = it0 + w; it < it0 + 64; it += 8) {
    const int n0 = it * 32;
    v16bf aq[2], bk[2];
#pragma unroll
    for (int mf = 0; mf < 2; ++mf) {
      const __bf16* rp = qb + (size_t)(mf * 16 + l15) * HW + n0;
      aq[mf] = load_afrag(rp, half);
#pragma unroll
      for (int i = 0; i < 16; ++i) { float x = b2f(aq[mf][i]); msq[mf] += x * x; }
    }
#pragma unroll
    for (int nf = 0; nf < 2; ++nf) {
      const __bf16* rp = kb + (size_t)(nf * 16 + l15) * HW + n0;
      bk[nf] = load_bfrag(rp, half);
#pragma unroll
      for (int i = 0; i < 16; ++i) { float x = b2f(bk[nf][i]); msk[nf] += x * x; }
    }
#pragma unroll
    for (int mf = 0; mf < 2; ++mf)
#pragma unroll
      for (int nf = 0; nf < 2; ++nf)
        g[mf][nf] = __builtin_amdgcn_wmma_f32_16x16x32_bf16(
            false, aq[mf], false, bk[nf], (short)0, g[mf][nf], false, false);
  }
  // cross-wave reduction into LDS
#pragma unroll
  for (int mf = 0; mf < 2; ++mf) {
    float s = msq[mf] + __shfl_xor(msq[mf], 16);
    if (lane < 16) atomicAdd(&sqq[mf * 16 + l15], s);
  }
#pragma unroll
  for (int nf = 0; nf < 2; ++nf) {
    float s = msk[nf] + __shfl_xor(msk[nf], 16);
    if (lane < 16) atomicAdd(&sqk[nf * 16 + l15], s);
  }
#pragma unroll
  for (int mf = 0; mf < 2; ++mf)
#pragma unroll
    for (int nf = 0; nf < 2; ++nf)
#pragma unroll
      for (int r = 0; r < 8; ++r)
        atomicAdd(&Gs[mf * 16 + r + half * 8][nf * 16 + l15], g[mf][nf][r]);
  __syncthreads();

  // flush partials to global
  float* Gg = Gws + (size_t)bh * 1024;
  for (int i = t; i < 1024; i += 256) atomicAdd(&Gg[i], ((float*)Gs)[i]);
  if (t < 32)       atomicAdd(&sqws[bh * 64 + t], sqq[t]);
  else if (t < 64)  atomicAdd(&sqws[bh * 64 + t], sqk[t - 32]);
}

// ---------------------------------------------------------------------------
// Kernel 3b: softmax with l2-norm + temperature folding; 64 blocks x 32 lanes.
// ---------------------------------------------------------------------------
__global__ __launch_bounds__(32) void xca_softmax(
    const float* __restrict__ Gws,          // [64][32][32]
    const float* __restrict__ sqws,         // [64][64]
    const float* __restrict__ temperature,  // [NHEADS]
    __bf16* __restrict__ attnw) {           // [64][32][32]
  const int bh = blockIdx.x;
  const int head = bh & 7;
  const int c = threadIdx.x;
  const float* G  = Gws + (size_t)bh * 1024;
  const float* sq = sqws + bh * 64;

  const float eps  = 1e-12f;
  const float temp = temperature[head];
  float nq = fmaxf(sqrtf(sq[c]), eps);
  float row[32];
  float mx = -3.4e38f;
  for (int d = 0; d < 32; ++d) {
    float nk = fmaxf(sqrtf(sq[32 + d]), eps);
    float v  = G[c * 32 + d] * temp / (nq * nk);
    row[d] = v;
    mx = fmaxf(mx, v);
  }
  float sum = 0.f;
  for (int d = 0; d < 32; ++d) { row[d] = __expf(row[d] - mx); sum += row[d]; }
  float r = 1.f / sum;
  for (int d = 0; d < 32; ++d) attnw[(size_t)bh * 1024 + c * 32 + d] = f2b(row[d] * r);
}

// ---------------------------------------------------------------------------
// Kernel 3c: out = attn(32x32) @ v(32xHW) over an n-slice; 512 blocks.
// K=32 -> single WMMA step; v transpose-staged through LDS.
// ---------------------------------------------------------------------------
__global__ __launch_bounds__(256) void xca_av(
    const __bf16* __restrict__ qkvc,   // [BATCH][768][HW]
    const __bf16* __restrict__ attnw,  // [64][32][32]
    __bf16* __restrict__ out) {        // [BATCH][256][HW]
  __shared__ __attribute__((aligned(16))) __bf16 vt[512 * 34];  // [n][d]

  const int t    = threadIdx.x;
  const int lane = t & 31;
  const int half = lane >> 4;
  const int l15  = lane & 15;
  const int w    = t >> 5;
  const int bh    = blockIdx.y;   // 0..63
  const int b     = bh >> 3;
  const int head  = bh & 7;
  const int slice = blockIdx.x;   // 0..7

  const __bf16* vb = qkvc + ((size_t)b * QKV_CH + 512 + head * 32) * HW;

  v16bf af[2];
#pragma unroll
  for (int mf = 0; mf < 2; ++mf)
    af[mf] = load_afrag(attnw + (size_t)bh * 1024 + (mf * 16 + l15) * 32, half);

  const int vd = t >> 3, vjb = (t & 7) * 64;
  const int nbeg = slice * 2048, nend = nbeg + 2048;
  for (int tile0 = nbeg; tile0 < nend; tile0 += 512) {
    // transpose-stage v[32][512] -> vt[512][32] (coalesced global reads)
    {
      const __bf16* src = vb + (size_t)vd * HW + tile0 + vjb;
#pragma unroll 8
      for (int j = 0; j < 64; ++j) vt[(vjb + j) * 34 + vd] = src[j];
    }
    __syncthreads();

    const int ncol = w * 64;
    v16bf bfr[4];
#pragma unroll
    for (int nf = 0; nf < 4; ++nf)
      bfr[nf] = load_bfrag(vt + (ncol + nf * 16 + l15) * 34, half);

    v8f o[2][4];
#pragma unroll
    for (int mf = 0; mf < 2; ++mf)
#pragma unroll
      for (int nf = 0; nf < 4; ++nf) {
#pragma unroll
        for (int i = 0; i < 8; ++i) o[mf][nf][i] = 0.f;
        o[mf][nf] = __builtin_amdgcn_wmma_f32_16x16x32_bf16(
            false, af[mf], false, bfr[nf], (short)0, o[mf][nf], false, false);
      }
#pragma unroll
    for (int mf = 0; mf < 2; ++mf)
#pragma unroll
      for (int nf = 0; nf < 4; ++nf)
#pragma unroll
        for (int r = 0; r < 8; ++r) {
          int ch  = head * 32 + mf * 16 + r + half * 8;
          int col = tile0 + ncol + nf * 16 + l15;
          out[((size_t)b * CDIM + ch) * HW + col] = f2b(o[mf][nf][r]);
        }
    __syncthreads();
  }
}

// ---------------------------------------------------------------------------
// Launch
// ---------------------------------------------------------------------------
extern "C" void kernel_launch(void* const* d_in, const int* in_sizes, int n_in,
                              void* d_out, int out_size, void* d_ws, size_t ws_size,
                              hipStream_t stream) {
  const float* x      = (const float*)d_in[0];  // [8][256][128][128]
  const float* w_qkv  = (const float*)d_in[1];  // [768][256]
  const float* w_dw   = (const float*)d_in[2];  // [768][1][3][3][3]
  const float* w_proj = (const float*)d_in[3];  // [256][256]
  const float* temp   = (const float*)d_in[4];  // [8][1][1]

  const size_t nqkv = (size_t)BATCH * QKV_CH * HW;   // 100,663,296
  const size_t nout = (size_t)BATCH * CDIM * HW;     //  33,554,432
  __bf16* qkv  = (__bf16*)d_ws;        // bf16 qkv (pre-conv)
  __bf16* qkvc = qkv + nqkv;           // bf16 qkv (post depthwise conv)
  __bf16* ao   = qkvc + nqkv;          // bf16 attention output
  float*  Gws  = (float*)((char*)d_ws + (2 * nqkv + nout) * sizeof(__bf16));
  float*  sqws = Gws + 64 * 1024;      // [64][64]
  __bf16* attnw = (__bf16*)(sqws + 64 * 64);
  (void)in_sizes; (void)n_in; (void)out_size; (void)ws_size;

  // 1) qkv = w_qkv @ x   (M=768, K=256, N=16384 per batch)
  gemm_channelmix<false, true><<<dim3(HW / 128, QKV_CH / 128, BATCH), 256, 0, stream>>>(
      w_qkv, (const void*)x, (void*)qkv, QKV_CH, CDIM);

  // 2) depthwise 3x3 (middle z-slice of w_dw)
  dwconv3x3<<<(unsigned)(nqkv / 256), 256, 0, stream>>>(qkv, w_dw, qkvc);

  // 3) transposed attention: zero accumulators, gram, softmax, attn@v
  zero_f32<<<(64 * 1024 + 64 * 64 + 255) / 256, 256, 0, stream>>>(Gws, 64 * 1024 + 64 * 64);
  xca_gram<<<dim3(8, 64), 256, 0, stream>>>(qkvc, Gws, sqws);
  xca_softmax<<<64, 32, 0, stream>>>(Gws, sqws, temp, attnw);
  xca_av<<<dim3(8, 64), 256, 0, stream>>>(qkvc, attnw, ao);

  // 4) out = w_proj @ attn_out  (M=256, K=256, N=16384 per batch)
  gemm_channelmix<true, false><<<dim3(HW / 128, CDIM / 128, BATCH), 256, 0, stream>>>(
      w_proj, (const void*)ao, d_out, CDIM, CDIM);
}
